// MultiHeadedAttention_43817256353960
// MI455X (gfx1250) — matrix operand order repaired and checked by reading, not compiled, not run on hardware
//
#include <hip/hip_runtime.h>

#define BDIM 2
#define SDIM 1536
#define DDIM 1024
#define HDIM 16
#define DKDIM 64
#define MDIM (BDIM * SDIM)   // 3072 rows for the big GEMMs
#define BHDIM (BDIM * HDIM)  // 32 (batch*heads)

typedef __attribute__((ext_vector_type(16))) __bf16 v16bf;
typedef __attribute__((ext_vector_type(8)))  float  v8f;

union FragB {
    v16bf bf;
    unsigned int u32[8];
    unsigned short u16[16];
};

__device__ __forceinline__ unsigned short f32_bf16(float f) {
    unsigned int u = __float_as_uint(f);
    u += 0x7FFFu + ((u >> 16) & 1u);   // round-to-nearest-even
    return (unsigned short)(u >> 16);
}

// 16-bit A fragment (16x32, MxK): lane r=lane%16 holds row r; half hs=lane/16.
// VGPR v -> K = 16*hs + 2v  (pairs K, K+1)
__device__ __forceinline__ int a_koff(int v, int hs) {
    return 16 * hs + 2 * v;
}
// 16-bit B fragment (32x16, KxN): lane holds column lane%16; K = 16*hs + 2v (pairs)
__device__ __forceinline__ int b_koff(int v, int hs) { return 16 * hs + 2 * v; }

// async global->LDS 16B copy (ASYNCcnt-tracked; per-lane addresses).
// Generic shared pointers carry the LDS byte offset in their low 32 bits
// (aperture is addr[63:32] per the ISA), so truncation yields the VDST value.
__device__ __forceinline__ void async_b128(const unsigned short* g,
                                           unsigned short* l) {
    unsigned lds = (unsigned)(size_t)l;
    unsigned long long ga = (unsigned long long)(size_t)g;
    asm volatile("global_load_async_to_lds_b128 %0, %1, off"
                 :: "v"(lds), "v"(ga) : "memory");
}

// ---------------------------------------------------------------- converts
__global__ void cvt_f32_to_bf16(const float* __restrict__ in,
                                unsigned short* __restrict__ out, int n) {
    int i = blockIdx.x * 256 + threadIdx.x;
    if (i < n) out[i] = f32_bf16(in[i]);
}

// counts[b,k] = sum_q mask1[b,0,q,k]
__global__ void colsum_kernel(const float* __restrict__ mask1,
                              float* __restrict__ counts) {
    int i = blockIdx.x * 256 + threadIdx.x;
    if (i >= BDIM * SDIM) return;
    int b = i / SDIM, k = i % SDIM;
    const float* p = mask1 + (size_t)b * SDIM * SDIM + k;
    float s = 0.f;
    for (int q = 0; q < SDIM; ++q) s += p[(size_t)q * SDIM];
    counts[i] = s;
}

// ---------------------------------------------------------------- tiled GEMM
// out[m,n] = sum_k A[m,k] * W[n,k] + bias[n]
// MODE 0: store bf16 into [B,H,S,DK] head layout   (Q/K projections)
// MODE 1: store f32 row-major [M,D]                (final projection)
// MODE 2: store bf16 transposed per head [B,H,DK,S] (V projection)
template <int MODE>
__global__ __launch_bounds__(256) void gemm_bf16_kernel(
    const unsigned short* __restrict__ A,    // [MDIM, DDIM] bf16
    const unsigned short* __restrict__ W,    // [DDIM, DDIM] bf16 (row n over k)
    const float* __restrict__ bias,          // [DDIM]
    void* __restrict__ out) {
    __shared__ __align__(16) unsigned short At[2][128 * 48]; // 128 rows x 32 k
    __shared__ __align__(16) unsigned short Bt[2][64 * 48];  // 64 W-rows x 32 k

    const int tid  = threadIdx.x;
    const int w    = tid >> 5;
    const int lane = tid & 31;
    const int hs   = lane >> 4;
    const int lr   = lane & 15;
    const int wm   = w >> 1;   // 0..3
    const int wn   = w & 1;    // 0..1
    const int m0   = blockIdx.y * 128;
    const int n0   = blockIdx.x * 64;

    // stage one 128x32 A tile + 64x32 B tile (3 async b128 per wave)
    auto stage = [&](int k0, int buf) {
        int i = tid;
#pragma unroll
        for (int r = 0; r < 2; ++r, i += 256) {
            int row = i >> 2, quad = i & 3;
            async_b128(A + (size_t)(m0 + row) * DDIM + k0 + quad * 8,
                       &At[buf][row * 48 + quad * 8]);
        }
        int row = tid >> 2, quad = tid & 3;
        async_b128(W + (size_t)(n0 + row) * DDIM + k0 + quad * 8,
                   &Bt[buf][row * 48 + quad * 8]);
    };

    v8f c[2][2];
    for (int tm = 0; tm < 2; ++tm)
        for (int tn = 0; tn < 2; ++tn) {
            float bv = bias[n0 + wn * 32 + tn * 16 + lr];
            for (int e = 0; e < 8; ++e) c[tm][tn][e] = bv;
        }

    stage(0, 0);
    int buf = 0;
    for (int k0 = 0; k0 < DDIM; k0 += 32) {
        if (k0 + 32 < DDIM) {
            stage(k0 + 32, buf ^ 1);
            asm volatile("s_wait_asynccnt 0x3" ::: "memory");
        } else {
            asm volatile("s_wait_asynccnt 0x0" ::: "memory");
        }
        __syncthreads();

        const unsigned int* A32 = (const unsigned int*)At[buf];
        const unsigned int* B32 = (const unsigned int*)Bt[buf];
        FragB a[2], bb[2];
#pragma unroll
        for (int tm = 0; tm < 2; ++tm) {
            int row = wm * 32 + tm * 16 + lr;
#pragma unroll
            for (int v = 0; v < 8; ++v)
                a[tm].u32[v] = A32[row * 24 + (a_koff(v, hs) >> 1)];
        }
#pragma unroll
        for (int tn = 0; tn < 2; ++tn) {
            int col = wn * 32 + tn * 16 + lr;
#pragma unroll
            for (int v = 0; v < 8; ++v)
                bb[tn].u32[v] = B32[col * 24 + (b_koff(v, hs) >> 1)];
        }
#pragma unroll
        for (int tm = 0; tm < 2; ++tm)
#pragma unroll
            for (int tn = 0; tn < 2; ++tn)
                c[tm][tn] = __builtin_amdgcn_wmma_f32_16x16x32_bf16(
                    false, a[tm].bf, false, bb[tn].bf, (short)0, c[tm][tn],
                    false, false);
        __syncthreads();
        buf ^= 1;
    }

#pragma unroll
    for (int tm = 0; tm < 2; ++tm) {
#pragma unroll
        for (int tn = 0; tn < 2; ++tn) {
            int n = n0 + wn * 32 + tn * 16 + lr;
#pragma unroll
            for (int v = 0; v < 8; ++v) {
                int m = m0 + wm * 32 + tm * 16 + v + 8 * hs;
                float val = c[tm][tn][v];
                int b = m / SDIM, s = m % SDIM;
                int hd = n >> 6, d = n & 63;
                if (MODE == 0) {
                    ((unsigned short*)out)[(((size_t)(b * HDIM + hd)) * SDIM + s) *
                                               DKDIM + d] = f32_bf16(val);
                } else if (MODE == 2) {
                    ((unsigned short*)out)[(((size_t)(b * HDIM + hd)) * DKDIM + d) *
                                               SDIM + s] = f32_bf16(val);
                } else {
                    ((float*)out)[(size_t)m * DDIM + n] = val;
                }
            }
        }
    }
}

// ---------------------------------------------------------------- scores
// scores[bh,q,k] = mask ? (QK^T/8) * (1 + mask1/count) : -1e9
__global__ __launch_bounds__(256) void scores_kernel(
    const unsigned short* __restrict__ Qh,  // [BH,S,DK] bf16
    const unsigned short* __restrict__ Kh,  // [BH,S,DK] bf16
    const int* __restrict__ mask,           // [B,S]
    const float* __restrict__ mask1,        // [B,S,S]
    const float* __restrict__ counts,       // [B,S]
    float* __restrict__ scores) {           // [BH,S,S]
    const int tid  = threadIdx.x;
    const int w    = tid >> 5;
    const int lane = tid & 31;
    const int hs   = lane >> 4;
    const int lr   = lane & 15;
    const int bh   = blockIdx.z;
    const int b    = bh >> 4;  // / HDIM
    const int q0   = (blockIdx.y * 8 + w) * 16;
    const int j0   = blockIdx.x * 64;

    v8f c[4];
    for (int t = 0; t < 4; ++t)
        for (int e = 0; e < 8; ++e) c[t][e] = 0.f;

#pragma unroll
    for (int d0 = 0; d0 < DKDIM; d0 += 32) {
        FragB a;
        const unsigned short* qp =
            Qh + ((size_t)bh * SDIM + q0 + lr) * DKDIM + d0;
#pragma unroll
        for (int v = 0; v < 8; ++v)
            a.u32[v] = *(const unsigned int*)(qp + a_koff(v, hs));
#pragma unroll
        for (int t = 0; t < 4; ++t) {
            FragB kb;
            const unsigned short* kp =
                Kh + ((size_t)bh * SDIM + j0 + t * 16 + lr) * DKDIM + d0;
#pragma unroll
            for (int v = 0; v < 8; ++v)
                kb.u32[v] = *(const unsigned int*)(kp + b_koff(v, hs));
            c[t] = __builtin_amdgcn_wmma_f32_16x16x32_bf16(
                false, a.bf, false, kb.bf, (short)0, c[t], false, false);
        }
    }

#pragma unroll
    for (int t = 0; t < 4; ++t) {
        int kc   = j0 + t * 16 + lr;
        float cn = counts[b * SDIM + kc];
        int mk   = mask[b * SDIM + kc];
        float iv = (cn > 0.f) ? __builtin_amdgcn_rcpf(cn) : 0.f;
#pragma unroll
        for (int v = 0; v < 8; ++v) {
            int q     = q0 + v + 8 * hs;
            float m1v = mask1[((size_t)b * SDIM + q) * SDIM + kc];
            float val = c[t][v] * 0.125f * (1.f + m1v * iv);
            if (mk == 0) val = -1e9f;
            scores[((size_t)bh * SDIM + q) * SDIM + kc] = val;
        }
    }
}

// ---------------------------------------------------------------- row stats
// One wave per score row: coalesced float4 loads cached in registers,
// shuffle-tree max, then sum of exp from registers.
__global__ __launch_bounds__(256) void rowstats_kernel(
    const float* __restrict__ scores, float* __restrict__ rmax,
    float* __restrict__ rinv) {
    const int w    = threadIdx.x >> 5;
    const int lane = threadIdx.x & 31;
    const size_t row = (size_t)blockIdx.x * 8 + w;  // over BH*S rows
    const float4* rp = (const float4*)(scores + row * SDIM);

    float4 v[12];
#pragma unroll
    for (int i = 0; i < 12; ++i) v[i] = rp[lane + 32 * i];

    float m = -3.4e38f;
#pragma unroll
    for (int i = 0; i < 12; ++i) {
        m = fmaxf(m, fmaxf(fmaxf(v[i].x, v[i].y), fmaxf(v[i].z, v[i].w)));
    }
#pragma unroll
    for (int off = 16; off >= 1; off >>= 1)
        m = fmaxf(m, __shfl_xor(m, off, 32));

    float s = 0.f;
#pragma unroll
    for (int i = 0; i < 12; ++i) {
        s += __expf(v[i].x - m) + __expf(v[i].y - m) + __expf(v[i].z - m) +
             __expf(v[i].w - m);
    }
#pragma unroll
    for (int off = 16; off >= 1; off >>= 1) s += __shfl_xor(s, off, 32);

    if (lane == 0) {
        rmax[row] = m;
        rinv[row] = __builtin_amdgcn_rcpf(s);
    }
}

// ---------------------------------------------------------------- P @ V
// Wave computes 16 query rows x 64 dims; P rebuilt from scores + row stats;
// V read from per-head transposed layout so B-fragments are contiguous.
__global__ __launch_bounds__(256) void softmax_pv_kernel(
    const float* __restrict__ scores,       // [BH,S,S]
    const unsigned short* __restrict__ Vt,  // [BH,DK,S] bf16
    const float* __restrict__ rmax,         // [BH*S]
    const float* __restrict__ rinv,         // [BH*S]
    unsigned short* __restrict__ xb) {      // [B,S,D] bf16
    const int tid  = threadIdx.x;
    const int w    = tid >> 5;
    const int lane = tid & 31;
    const int hs   = lane >> 4;
    const int lr   = lane & 15;
    const int bh   = blockIdx.y;
    const int b    = bh >> 4;
    const int hd   = bh & 15;
    const int q0   = (blockIdx.x * 8 + w) * 16;

    const float mx = rmax[(size_t)bh * SDIM + q0 + lr];
    const float iv = rinv[(size_t)bh * SDIM + q0 + lr];
    const float* sp = scores + ((size_t)bh * SDIM + q0 + lr) * SDIM;

    v8f c[4];
    for (int t = 0; t < 4; ++t)
        for (int e = 0; e < 8; ++e) c[t][e] = 0.f;

    for (int j0 = 0; j0 < SDIM; j0 += 32) {
        FragB p;
#pragma unroll
        for (int v = 0; v < 8; ++v) {
            int k = j0 + a_koff(v, hs);
            float2 s01 = *(const float2*)(sp + k);
            float p0 = __expf(s01.x - mx) * iv;
            float p1 = __expf(s01.y - mx) * iv;
            p.u32[v] = (unsigned int)f32_bf16(p0) |
                       ((unsigned int)f32_bf16(p1) << 16);
        }
#pragma unroll
        for (int t = 0; t < 4; ++t) {
            FragB vb;
            const unsigned short* vp =
                Vt + ((size_t)(bh * DKDIM + t * 16 + lr)) * SDIM + j0 + 16 * hs;
#pragma unroll
            for (int v = 0; v < 8; ++v)
                vb.u32[v] = *(const unsigned int*)(vp + 2 * v);
            c[t] = __builtin_amdgcn_wmma_f32_16x16x32_bf16(
                false, p.bf, false, vb.bf, (short)0, c[t], false, false);
        }
    }

#pragma unroll
    for (int t = 0; t < 4; ++t) {
        int d = t * 16 + lr;
#pragma unroll
        for (int v = 0; v < 8; ++v) {
            int q = q0 + v + 8 * hs;
            xb[((size_t)(b * SDIM + q)) * DDIM + hd * DKDIM + d] =
                f32_bf16(c[t][v]);
        }
    }
}

// ---------------------------------------------------------------- launcher
extern "C" void kernel_launch(void* const* d_in, const int* in_sizes, int n_in,
                              void* d_out, int out_size, void* d_ws,
                              size_t ws_size, hipStream_t stream) {
    (void)in_sizes; (void)n_in; (void)out_size; (void)ws_size;

    const float* query = (const float*)d_in[0];
    const float* keyi  = (const float*)d_in[1];
    const float* value = (const float*)d_in[2];
    const int*   mask  = (const int*)d_in[3];
    const float* mask1 = (const float*)d_in[4];
    const float* Wq = (const float*)d_in[5];
    const float* bq = (const float*)d_in[6];
    const float* Wk = (const float*)d_in[7];
    const float* bk = (const float*)d_in[8];
    const float* Wv = (const float*)d_in[9];
    const float* bv = (const float*)d_in[10];
    const float* Wo = (const float*)d_in[11];
    const float* bo = (const float*)d_in[12];

    float* ret    = (float*)d_out;
    float* scores = ret + (size_t)BDIM * SDIM * DDIM;

    char* ws = (char*)d_ws;
    size_t o = 0;
    auto alloc = [&](size_t bytes) {
        void* p = ws + o;
        o += (bytes + 255) & ~(size_t)255;
        return p;
    };
    unsigned short* bufIn = (unsigned short*)alloc((size_t)MDIM * DDIM * 2);
    unsigned short* bufW  = (unsigned short*)alloc((size_t)DDIM * DDIM * 2);
    unsigned short* Qh = (unsigned short*)alloc((size_t)BHDIM * SDIM * DKDIM * 2);
    unsigned short* Kh = (unsigned short*)alloc((size_t)BHDIM * SDIM * DKDIM * 2);
    unsigned short* Vt = (unsigned short*)alloc((size_t)BHDIM * SDIM * DKDIM * 2);
    float* counts      = (float*)alloc((size_t)BDIM * SDIM * 4);
    float* rmaxb       = (float*)alloc((size_t)BHDIM * SDIM * 4);
    float* rinvb       = (float*)alloc((size_t)BHDIM * SDIM * 4);
    unsigned short* xb = (unsigned short*)alloc((size_t)MDIM * DDIM * 2);

    const int nIn = MDIM * DDIM;      // 3.1M
    const int nW  = DDIM * DDIM;      // 1.0M
    const dim3 gIn((nIn + 255) / 256), gW((nW + 255) / 256), blk(256);
    const dim3 gGemm(DDIM / 64, MDIM / 128);          // (16, 24)
    const dim3 gScore(SDIM / 64, SDIM / 128, BHDIM);  // (24, 12, 32)
    const dim3 gStats((BHDIM * SDIM) / 8);            // 6144
    const dim3 gSpv(SDIM / 128, BHDIM);               // (12, 32)

    // Q projection
    cvt_f32_to_bf16<<<gIn, blk, 0, stream>>>(query, bufIn, nIn);
    cvt_f32_to_bf16<<<gW, blk, 0, stream>>>(Wq, bufW, nW);
    gemm_bf16_kernel<0><<<gGemm, blk, 0, stream>>>(bufIn, bufW, bq, (void*)Qh);
    // K projection
    cvt_f32_to_bf16<<<gIn, blk, 0, stream>>>(keyi, bufIn, nIn);
    cvt_f32_to_bf16<<<gW, blk, 0, stream>>>(Wk, bufW, nW);
    gemm_bf16_kernel<0><<<gGemm, blk, 0, stream>>>(bufIn, bufW, bk, (void*)Kh);
    // V projection (stored transposed per head)
    cvt_f32_to_bf16<<<gIn, blk, 0, stream>>>(value, bufIn, nIn);
    cvt_f32_to_bf16<<<gW, blk, 0, stream>>>(Wv, bufW, nW);
    gemm_bf16_kernel<2><<<gGemm, blk, 0, stream>>>(bufIn, bufW, bv, (void*)Vt);

    // mask1 column counts (collapsed axis=-2 softmax)
    colsum_kernel<<<(BDIM * SDIM + 255) / 256, blk, 0, stream>>>(mask1, counts);

    // scores = masked, scaled QK^T  (also a required output)
    scores_kernel<<<gScore, blk, 0, stream>>>(Qh, Kh, mask, mask1, counts,
                                              scores);

    // per-row max / inverse-sum
    rowstats_kernel<<<gStats, blk, 0, stream>>>(scores, rmaxb, rinvb);

    // softmax over k + P @ V -> xb (bf16, [B,S,D])
    softmax_pv_kernel<<<gSpv, blk, 0, stream>>>(scores, Vt, rmaxb, rinvb, xb);

    // output projection -> ret (f32)
    cvt_f32_to_bf16<<<gW, blk, 0, stream>>>(Wo, bufW, nW);
    gemm_bf16_kernel<1><<<gGemm, blk, 0, stream>>>(xb, bufW, bo, (void*)ret);
}
